// ContinuousGraphDiffusion_5420248728015
// MI455X (gfx1250) — compile-verified, matrix operand order checked by reading, not run on hardware
//
#include <hip/hip_runtime.h>
#include <hip/hip_bf16.h>
#include <math.h>

#define N_NODES 512
#define H_DIM   256
#define L_LAYERS 4
#define T_STEPS 100
#define B_BATCH 32
#define NE_EDGES ((long long)N_NODES*(N_NODES-1)/2)   // 130816

typedef __attribute__((ext_vector_type(16))) __bf16 v16bf;
typedef __attribute__((ext_vector_type(8)))  float  v8f;

__device__ __forceinline__ __bf16 to_bf(float f) {
    unsigned u = __builtin_bit_cast(unsigned, f);
    unsigned r = (u + 0x7FFFu + ((u >> 16) & 1u)) >> 16;
    unsigned short s = (unsigned short)r;
    return __builtin_bit_cast(__bf16, s);
}

// ---------------------------------------------------------------- schedule
__global__ void k_init(float* sab, float* somab, float* lossacc) {
    if (threadIdx.x == 0 && blockIdx.x == 0) {
        const double s = 0.008;
        const double PI = 3.14159265358979323846;
        double c0 = cos((0.0 / T_STEPS + s) / (1.0 + s) * PI * 0.5);
        double ab0 = c0 * c0;
        double abPrev = 1.0;
        float abar = 1.0f;           // f32 cumprod (matches np.cumprod on f32)
        for (int i = 1; i <= T_STEPS; ++i) {
            double c = cos(((double)i / T_STEPS + s) / (1.0 + s) * PI * 0.5);
            double ab = c * c / ab0;
            double beta = 1.0 - ab / abPrev;
            if (beta > 0.999) beta = 0.999;
            abPrev = ab;
            float alphaf = 1.0f - (float)beta;
            abar *= alphaf;
            sab[i - 1]   = sqrtf(abar);
            somab[i - 1] = sqrtf(1.0f - abar);
        }
        lossacc[0] = 0.0f;
    }
}

// ---------------------------------------------------------------- f32 [K x N] -> bf16 transposed [N x K]
__global__ void k_convT(const float* __restrict__ src, __bf16* __restrict__ dst,
                        int Kd, int Nd) {
    int idx = blockIdx.x * 256 + threadIdx.x;
    if (idx < Kd * Nd) {
        int k = idx / Nd, n = idx % Nd;
        dst[(size_t)n * Kd + k] = to_bf(src[idx]);
    }
}

// ---------------------------------------------------------------- adj_t / adj_norm
__global__ void k_prep(const float* __restrict__ adj0, const float* __restrict__ noise,
                       const int* __restrict__ t, const float* __restrict__ sab,
                       const float* __restrict__ somab,
                       __bf16* __restrict__ adjT, __bf16* __restrict__ adjN) {
    int i = blockIdx.x, b = blockIdx.y, tid = threadIdx.x;
    int tb = t[b];
    float a = sab[tb], o = somab[tb];
    size_t rowOff = ((size_t)b * N_NODES + i) * N_NODES;
    float x0 = a * adj0[rowOff + tid]       + o * noise[rowOff + tid];
    float x1 = a * adj0[rowOff + tid + 256] + o * noise[rowOff + tid + 256];
    __shared__ float red[256];
    red[tid] = fabsf(x0) + fabsf(x1);
    __syncthreads();
    for (int s = 128; s > 0; s >>= 1) {
        if (tid < s) red[tid] += red[tid + s];
        __syncthreads();
    }
    float inv = 1.0f / (red[0] + 1.0f);
    adjT[rowOff + tid]       = to_bf(x0);
    adjT[rowOff + tid + 256] = to_bf(x1);
    adjN[rowOff + tid]       = to_bf(x0 * inv);
    adjN[rowOff + tid + 256] = to_bf(x1 * inv);
}

// ---------------------------------------------------------------- WMMA GEMM
__device__ __forceinline__ v16bf load_frag(const __bf16* p) {
    v16bf f;
#pragma unroll
    for (int i = 0; i < 8; ++i) { f[i] = p[i]; f[i + 8] = p[i + 16]; }
    return f;
}

__device__ __forceinline__ void async_b128(unsigned lds_off, const void* g) {
    asm volatile("global_load_async_to_lds_b128 %0, %1, off"
                 :: "v"(lds_off), "v"(g) : "memory");
}
__device__ __forceinline__ void wait_async0() {
    asm volatile("s_wait_asynccnt 0x0" ::: "memory");
}

// C[bz](M x N) = A[bz](M x K) * BT[bz](N x K)^T  [+ C][+ bias]
// block tile 128x64, 8 waves, wave tile 32x32 (4 WMMA / K-step), K step 32.
// Double-buffered LDS fed by global_load_async_to_lds_b128 (ASYNCcnt pipeline).
template <bool HAS_BIAS, bool ACCUM, bool WF32, bool WBF, bool WBFT>
__global__ void __launch_bounds__(256)
k_gemm_bf16(const __bf16* __restrict__ A, int lda, long long strideA,
            const __bf16* __restrict__ BT, int ldbt, long long strideB,
            float* __restrict__ Cf, __bf16* __restrict__ Cbf,
            __bf16* __restrict__ CbfT,
            int ldc, int ldct, long long strideC,
            const float* __restrict__ bias, int K) {
    __shared__ __bf16 lA[2][128 * 32];   // [buf][m][k]
    __shared__ __bf16 lB[2][64 * 32];    // [buf][n][k]

    int bn = blockIdx.x, bm = blockIdx.y, bz = blockIdx.z;
    A  += (size_t)bz * strideA;
    BT += (size_t)bz * strideB;
    if (WF32 || ACCUM) Cf   += (size_t)bz * strideC;
    if (WBF)           Cbf  += (size_t)bz * strideC;
    if (WBFT)          CbfT += (size_t)bz * strideC;

    int tid  = threadIdx.x;
    int lane = tid & 31;
    int wave = tid >> 5;
    int m0 = (wave & 3) * 32;            // 4 M-tiles of 32 -> 128
    int n0 = (wave >> 2) * 32;           // 2 N-tiles of 32 -> 64
    int lrow  = lane & 15;
    int kbase = (lane & 16) ? 8 : 0;     // lanes 16-31 hold K=8..15 / 24..31

    // staging: A = 512 chunks of 8 bf16 (2/thread), B = 256 chunks (1/thread)
    int sRow = tid >> 2, sCC = tid & 3;
    unsigned aOff0 = (unsigned)(uintptr_t)&lA[0][(sRow)      * 32 + sCC * 8];
    unsigned aOff1 = (unsigned)(uintptr_t)&lA[0][(sRow + 64) * 32 + sCC * 8];
    unsigned bOff  = (unsigned)(uintptr_t)&lB[0][(sRow)      * 32 + sCC * 8];
    const __bf16* Abase = A  + (size_t)(bm * 128 + sRow) * lda  + sCC * 8;
    const __bf16* Abase2 = Abase + (size_t)64 * lda;
    const __bf16* Bbase = BT + (size_t)(bn * 64  + sRow) * ldbt + sCC * 8;

    const unsigned ABUF = sizeof(__bf16) * 128 * 32;   // 8192
    const unsigned BBUF = sizeof(__bf16) * 64 * 32;    // 4096

    auto issue_tile = [&](int buf, int kt) {
        async_b128(aOff0 + buf * ABUF, Abase  + kt);
        async_b128(aOff1 + buf * ABUF, Abase2 + kt);
        async_b128(bOff  + buf * BBUF, Bbase  + kt);
    };

    v8f acc[2][2] = {{{}, {}}, {{}, {}}};

    issue_tile(0, 0);
    wait_async0();
    __syncthreads();

    int cur = 0;
    for (int kt = 0; kt < K; kt += 32) {
        int nxt = cur ^ 1;
        if (kt + 32 < K) issue_tile(nxt, kt + 32);   // overlap fetch with math

        v16bf a0 = load_frag(&lA[cur][(m0      + lrow) * 32 + kbase]);
        v16bf a1 = load_frag(&lA[cur][(m0 + 16 + lrow) * 32 + kbase]);
        v16bf b0 = load_frag(&lB[cur][(n0      + lrow) * 32 + kbase]);
        v16bf b1 = load_frag(&lB[cur][(n0 + 16 + lrow) * 32 + kbase]);
        acc[0][0] = __builtin_amdgcn_wmma_f32_16x16x32_bf16(false, a0, false, b0,
                                                            (short)0, acc[0][0], false, false);
        acc[0][1] = __builtin_amdgcn_wmma_f32_16x16x32_bf16(false, a0, false, b1,
                                                            (short)0, acc[0][1], false, false);
        acc[1][0] = __builtin_amdgcn_wmma_f32_16x16x32_bf16(false, a1, false, b0,
                                                            (short)0, acc[1][0], false, false);
        acc[1][1] = __builtin_amdgcn_wmma_f32_16x16x32_bf16(false, a1, false, b1,
                                                            (short)0, acc[1][1], false, false);

        wait_async0();
        __syncthreads();
        cur = nxt;
    }

    int moff = (lane & 16) ? 8 : 0;      // D: lanes 16-31 hold rows r+8
#pragma unroll
    for (int q = 0; q < 4; ++q) {
        int gmb = bm * 128 + m0 + (q >> 1) * 16 + moff;
        int gn  = bn * 64  + n0 + (q & 1)  * 16 + lrow;
        float bv = 0.0f;
        if (HAS_BIAS) bv = bias[gn];
        const v8f& av = acc[q >> 1][q & 1];
#pragma unroll
        for (int r = 0; r < 8; ++r) {
            int gm = gmb + r;
            int idx = gm * ldc + gn;               // < 2^31 in-batch
            float v = av[r];
            if (ACCUM)    v += Cf[idx];
            if (HAS_BIAS) v += bv;
            if (WF32) Cf[idx]  = v;
            if (WBF)  Cbf[idx] = to_bf(v);
            if (WBFT) CbfT[gn * ldct + gm] = to_bf(v);
        }
    }
}

// ---------------------------------------------------------------- t embedding
__global__ void k_temb(const int* __restrict__ t, const float* __restrict__ W1,
                       const float* __restrict__ b1, const float* __restrict__ W2,
                       const float* __restrict__ b2, float* __restrict__ temb) {
    int b = blockIdx.x, j = threadIdx.x;
    float tf = (float)t[b] / (float)T_STEPS;
    __shared__ float s[H_DIM];
    float x = tf * W1[j] + b1[j];
    s[j] = x / (1.0f + __expf(-x));
    __syncthreads();
    float acc = b2[j];
    for (int k = 0; k < H_DIM; ++k) acc += s[k] * W2[k * H_DIM + j];
    temb[b * H_DIM + j] = acc;
}

__global__ void k_tbias(const float* __restrict__ temb, const float* __restrict__ tpW,
                        const float* __restrict__ tpb, float* __restrict__ tbias) {
    int b = blockIdx.x, l = blockIdx.y, j = threadIdx.x;
    __shared__ float s[H_DIM];
    s[j] = temb[b * H_DIM + j];
    __syncthreads();
    const float* W = tpW + (size_t)l * H_DIM * H_DIM;
    float acc = tpb[l * H_DIM + j];
    for (int k = 0; k < H_DIM; ++k) acc += s[k] * W[k * H_DIM + j];
    tbias[((size_t)l * B_BATCH + b) * H_DIM + j] = acc;
}

// ---------------------------------------------------------------- residual + LN
__global__ void k_ln(const float* __restrict__ hn, const float* __restrict__ updb,
                     const float* __restrict__ tbias, const float* __restrict__ g,
                     const float* __restrict__ be,
                     float* __restrict__ h, __bf16* __restrict__ hbf) {
    int i = blockIdx.x, b = blockIdx.y, j = threadIdx.x;
    size_t idx = ((size_t)b * N_NODES + i) * H_DIM + j;
    float x = hn[idx] + updb[j] + tbias[b * H_DIM + j] + h[idx];
    __shared__ float r1[256], r2[256];
    r1[j] = x; r2[j] = x * x;
    __syncthreads();
    for (int s = 128; s > 0; s >>= 1) {
        if (j < s) { r1[j] += r1[j + s]; r2[j] += r2[j + s]; }
        __syncthreads();
    }
    float mu  = r1[0] * (1.0f / H_DIM);
    float var = r2[0] * (1.0f / H_DIM) - mu * mu;
    float y = (x - mu) * rsqrtf(var + 1e-5f) * g[j] + be[j];
    h[idx] = y;
    hbf[idx] = to_bf(y);
}

// ---------------------------------------------------------------- node sum + head
__global__ void k_hgraph(const float* __restrict__ h, float* __restrict__ hg) {
    int b = blockIdx.x, j = threadIdx.x;
    const float* p = h + (size_t)b * N_NODES * H_DIM + j;
    float acc = 0.0f;
    for (int i = 0; i < N_NODES; ++i) acc += p[(size_t)i * H_DIM];
    hg[b * H_DIM + j] = acc;
}

__global__ void k_h2(const float* __restrict__ hg, const float* __restrict__ W1,
                     const float* __restrict__ b1, float* __restrict__ h2) {
    int b = blockIdx.x, j = threadIdx.x;
    __shared__ float s[H_DIM];
    s[j] = hg[b * H_DIM + j];
    __syncthreads();
    float acc = b1[j];
    for (int k = 0; k < H_DIM; ++k) acc += s[k] * W1[k * H_DIM + j];
    h2[b * H_DIM + j] = acc / (1.0f + __expf(-acc));
}

// ---------------------------------------------------------------- fused proj + MSE
__device__ __forceinline__ long long triOff(int i) {
    return (long long)i * (2 * N_NODES - i - 1) / 2;
}

__global__ void k_loss(const float* __restrict__ h2, const float* __restrict__ W2,
                       const float* __restrict__ b2, const float* __restrict__ noise,
                       float* __restrict__ acc) {
    int tid = threadIdx.x, wave = tid >> 5, lane = tid & 31;   // lane == batch
    long long e = (long long)blockIdx.x * 8 + wave;
    double disc = (double)(2 * N_NODES - 1) * (2 * N_NODES - 1) - 8.0 * (double)e;
    int i = (int)(((double)(2 * N_NODES - 1) - sqrt(disc)) * 0.5);
    while (triOff(i + 1) <= e) ++i;
    while (triOff(i) > e) --i;
    int j = i + 1 + (int)(e - triOff(i));

    const float* hb = h2 + lane * H_DIM;
    float p = b2[e];
    for (int k = 0; k < H_DIM; ++k) p += hb[k] * W2[(size_t)k * NE_EDGES + e];
    float d = p - noise[((size_t)lane * N_NODES + i) * N_NODES + j];
    __shared__ float red[256];
    red[tid] = d * d;
    __syncthreads();
    for (int s = 128; s > 0; s >>= 1) {
        if (tid < s) red[tid] += red[tid + s];
        __syncthreads();
    }
    if (tid == 0) atomicAdd(acc, red[0]);
}

__global__ void k_final(const float* __restrict__ acc, float* __restrict__ out) {
    if (threadIdx.x == 0 && blockIdx.x == 0)
        out[0] = acc[0] / (float)((long long)B_BATCH * NE_EDGES);
}

// ================================================================ launch
extern "C" void kernel_launch(void* const* d_in, const int* in_sizes, int n_in,
                              void* d_out, int out_size, void* d_ws, size_t ws_size,
                              hipStream_t stream) {
    const float* adj_0  = (const float*)d_in[0];
    const float* noise  = (const float*)d_in[1];
    const float* te_W1  = (const float*)d_in[2];
    const float* te_b1  = (const float*)d_in[3];
    const float* te_W2  = (const float*)d_in[4];
    const float* te_b2  = (const float*)d_in[5];
    const float* ip_W   = (const float*)d_in[6];
    const float* ip_b   = (const float*)d_in[7];
    const float* msg_W  = (const float*)d_in[8];
    const float* msg_b  = (const float*)d_in[9];
    const float* upd_W  = (const float*)d_in[10];
    const float* upd_b  = (const float*)d_in[11];
    const float* ln_g   = (const float*)d_in[12];
    const float* ln_b   = (const float*)d_in[13];
    const float* tp_W   = (const float*)d_in[14];
    const float* tp_b   = (const float*)d_in[15];
    const float* op_W1  = (const float*)d_in[16];
    const float* op_b1  = (const float*)d_in[17];
    const float* op_W2  = (const float*)d_in[18];
    const float* op_b2  = (const float*)d_in[19];
    const int*   t      = (const int*)d_in[20];

    char* ws = (char*)d_ws;
    size_t off = 0;
    auto alloc = [&](size_t bytes) -> void* {
        void* p = ws + off;
        off = (off + bytes + 255) & ~(size_t)255;
        return p;
    };

    float*  sab     = (float*)alloc(T_STEPS * 4);
    float*  somab   = (float*)alloc(T_STEPS * 4);
    float*  lacc    = (float*)alloc(4);
    float*  temb    = (float*)alloc((size_t)B_BATCH * H_DIM * 4);
    float*  tbias   = (float*)alloc((size_t)L_LAYERS * B_BATCH * H_DIM * 4);
    float*  hgraph  = (float*)alloc((size_t)B_BATCH * H_DIM * 4);
    float*  h2v     = (float*)alloc((size_t)B_BATCH * H_DIM * 4);
    __bf16* ipWT    = (__bf16*)alloc((size_t)N_NODES * H_DIM * 2);              // [256 x 512]
    __bf16* msgWT   = (__bf16*)alloc((size_t)L_LAYERS * H_DIM * H_DIM * 2);     // [256 x 256] x4
    __bf16* updWT   = (__bf16*)alloc((size_t)L_LAYERS * 2 * H_DIM * H_DIM * 2); // [256 x 512] x4
    __bf16* adjT    = (__bf16*)alloc((size_t)B_BATCH * N_NODES * N_NODES * 2);
    __bf16* adjN    = (__bf16*)alloc((size_t)B_BATCH * N_NODES * N_NODES * 2);
    float*  h_f32   = (float*) alloc((size_t)B_BATCH * N_NODES * H_DIM * 4);
    __bf16* h_bf    = (__bf16*)alloc((size_t)B_BATCH * N_NODES * H_DIM * 2);
    __bf16* hmT     = (__bf16*)alloc((size_t)B_BATCH * N_NODES * H_DIM * 2);    // [256 x 512] per batch
    __bf16* msg_bf  = (__bf16*)alloc((size_t)B_BATCH * N_NODES * H_DIM * 2);
    float*  hn_f32  = (float*) alloc((size_t)B_BATCH * N_NODES * H_DIM * 4);

    const long long sAdj = (long long)N_NODES * N_NODES;
    const long long sH   = (long long)N_NODES * H_DIM;

    k_init<<<1, 64, 0, stream>>>(sab, somab, lacc);

    // transposed bf16 weights
    k_convT<<<(N_NODES * H_DIM + 255) / 256, 256, 0, stream>>>(ip_W, ipWT, N_NODES, H_DIM);
    for (int l = 0; l < L_LAYERS; ++l) {
        k_convT<<<(H_DIM * H_DIM + 255) / 256, 256, 0, stream>>>(
            msg_W + (size_t)l * H_DIM * H_DIM, msgWT + (size_t)l * H_DIM * H_DIM, H_DIM, H_DIM);
        k_convT<<<(2 * H_DIM * H_DIM + 255) / 256, 256, 0, stream>>>(
            upd_W + (size_t)l * 2 * H_DIM * H_DIM, updWT + (size_t)l * 2 * H_DIM * H_DIM,
            2 * H_DIM, H_DIM);
    }

    k_prep<<<dim3(N_NODES, B_BATCH), 256, 0, stream>>>(adj_0, noise, t, sab, somab, adjT, adjN);

    k_temb<<<B_BATCH, H_DIM, 0, stream>>>(t, te_W1, te_b1, te_W2, te_b2, temb);
    k_tbias<<<dim3(B_BATCH, L_LAYERS), H_DIM, 0, stream>>>(temb, tp_W, tp_b, tbias);

    dim3 gemmGrid(H_DIM / 64, N_NODES / 128, B_BATCH);   // (4, 4, 32)

    // h0 = adj_t @ ip_W + ip_b              (K = 512): bias, write f32 + bf16
    k_gemm_bf16<true, false, true, true, false><<<gemmGrid, 256, 0, stream>>>(
        adjT, N_NODES, sAdj, ipWT, N_NODES, 0,
        h_f32, h_bf, nullptr, H_DIM, 0, sH, ip_b, N_NODES);

    for (int l = 0; l < L_LAYERS; ++l) {
        // hm^T = (h @ msg_W[l] + msg_b[l])^T  (K = 256): bias, write bf16 transposed
        k_gemm_bf16<true, false, false, false, true><<<gemmGrid, 256, 0, stream>>>(
            h_bf, H_DIM, sH, msgWT + (size_t)l * H_DIM * H_DIM, H_DIM, 0,
            nullptr, nullptr, hmT, 0, N_NODES, sH, msg_b + l * H_DIM, H_DIM);
        // msg = adj_norm @ hm                 (K = 512): write bf16
        k_gemm_bf16<false, false, false, true, false><<<gemmGrid, 256, 0, stream>>>(
            adjN, N_NODES, sAdj, hmT, N_NODES, sH,
            nullptr, msg_bf, nullptr, H_DIM, 0, sH, nullptr, N_NODES);
        // hn = h @ upd_W[l][:256]             (K = 256): write f32
        k_gemm_bf16<false, false, true, false, false><<<gemmGrid, 256, 0, stream>>>(
            h_bf, H_DIM, sH, updWT + (size_t)l * 2 * H_DIM * H_DIM, 2 * H_DIM, 0,
            hn_f32, nullptr, nullptr, H_DIM, 0, sH, nullptr, H_DIM);
        // hn += msg @ upd_W[l][256:]          (K = 256): accumulate + write f32
        k_gemm_bf16<false, true, true, false, false><<<gemmGrid, 256, 0, stream>>>(
            msg_bf, H_DIM, sH, updWT + (size_t)l * 2 * H_DIM * H_DIM + H_DIM, 2 * H_DIM, 0,
            hn_f32, nullptr, nullptr, H_DIM, 0, sH, nullptr, H_DIM);
        // h = LN(hn + upd_b + t_bias + h)
        k_ln<<<dim3(N_NODES, B_BATCH), H_DIM, 0, stream>>>(hn_f32,
                                                           upd_b + l * H_DIM,
                                                           tbias + (size_t)l * B_BATCH * H_DIM,
                                                           ln_g + l * H_DIM, ln_b + l * H_DIM,
                                                           h_f32, h_bf);
    }

    k_hgraph<<<B_BATCH, H_DIM, 0, stream>>>(h_f32, hgraph);
    k_h2<<<B_BATCH, H_DIM, 0, stream>>>(hgraph, op_W1, op_b1, h2v);

    k_loss<<<(unsigned)(NE_EDGES / 8), 256, 0, stream>>>(h2v, op_W2, op_b2, noise, lacc);
    k_final<<<1, 64, 0, stream>>>(lacc, (float*)d_out);
}
